// GraphLearner_54949811585481
// MI455X (gfx1250) — matrix operand order repaired and checked by reading, not compiled, not run on hardware
//
#include <hip/hip_runtime.h>
#include <hip/hip_bf16.h>

typedef __attribute__((ext_vector_type(16))) _Float16 v16h;
typedef __attribute__((ext_vector_type(8)))  _Float16 v8h;
typedef __attribute__((ext_vector_type(8)))  float    v8f;

#define B_   32
#define C_   1000
#define H_   512
#define N_   1001
#define NPAD 1024
#define NTILES 64    // NPAD/16
#define KTILES 16    // H/32
#define LDSROW 520   // halves per LDS tile row (512 + 8 pad -> 4-bank shift/row)

// ---------------------------------------------------------------------------
// K1: per-node L2 normalize (over H), write f16 features, zero-pad to NPAD rows
// ---------------------------------------------------------------------------
__global__ void __launch_bounds__(256)
k_normalize(const float* __restrict__ in_text, const float* __restrict__ in_img,
            const float* __restrict__ base_text, const float* __restrict__ base_img,
            _Float16* __restrict__ Fh) {
  int row = blockIdx.x;            // bm*NPAD + n
  int bm  = row >> 10;             // /NPAD
  int n   = row & (NPAD - 1);
  int b   = bm >> 1, mod = bm & 1;
  _Float16* dst = Fh + (size_t)row * H_;
  int t = threadIdx.x;
  if (n >= N_) {                   // padded rows -> zeros
    dst[t] = (_Float16)0.f;
    dst[t + 256] = (_Float16)0.f;
    return;
  }
  const float* src;
  if (n == 0) src = (mod ? in_img : in_text) + (size_t)b * H_;
  else        src = (mod ? base_img : base_text) + ((size_t)b * C_ + (n - 1)) * H_;
  float x0 = src[t], x1 = src[t + 256];
  __shared__ float red[256];
  red[t] = x0 * x0 + x1 * x1;
  __syncthreads();
  for (int s = 128; s > 0; s >>= 1) {
    if (t < s) red[t] += red[t + s];
    __syncthreads();
  }
  float inv = 1.0f / fmaxf(sqrtf(red[0]), 1e-12f);
  dst[t]       = (_Float16)(x0 * inv);
  dst[t + 256] = (_Float16)(x1 * inv);
}

// ---------------------------------------------------------------------------
// K2: Gram rowsums via WMMA. 8 waves/workgroup (8 m-tiles), B tiles staged
// through double-buffered LDS; A fragments resident in registers.
// ---------------------------------------------------------------------------
__global__ void __launch_bounds__(256)
k_gram(const _Float16* __restrict__ Fh,
       float* __restrict__ rowsum, float* __restrict__ a0row) {
  const int tid  = threadIdx.x;
  const int wave = tid >> 5;                 // 0..7
  const int lane = tid & 31;
  const int bm   = blockIdx.y;               // 0..63
  const int mt   = (blockIdx.x << 3) + wave; // 0..63

  const _Float16* F = Fh + (size_t)bm * NPAD * H_;

  // ---- A fragments for this wave's 16 rows, full K=512, resident in VGPRs.
  // A k order used here (contiguous, not the ISA table order): lane<16 -> K base 0, lane>=16 -> K base 16;
  // frag elems 0..15 = K base..base+15 (contiguous, matches B).
  const int mrow  = (mt << 4) + (lane & 15);
  const int abase = (lane >> 4) << 4;
  const _Float16* arow = F + (size_t)mrow * H_;
  union AF { v16h v; v8h h[2]; };
  AF a[KTILES];
#pragma unroll
  for (int kt = 0; kt < KTILES; ++kt) {
    a[kt].h[0] = *(const v8h*)(arow + kt * 32 + abase);
    a[kt].h[1] = *(const v8h*)(arow + kt * 32 + abase + 8);
  }

  __shared__ __align__(16) _Float16 lds[2][16 * LDSROW];

  // staging: 4 x 16B chunks per thread; chunk ci -> row = ci>>6, col8 = ci&63
  v8h st[4];
#pragma unroll
  for (int j = 0; j < 4; ++j) {               // prefetch n-tile 0
    const int ci = tid + (j << 8);
    st[j] = *(const v8h*)(F + (size_t)(ci >> 6) * H_ + ((ci & 63) << 3));
  }
#pragma unroll
  for (int j = 0; j < 4; ++j) {
    const int ci = tid + (j << 8);
    *(v8h*)(&lds[0][(ci >> 6) * LDSROW + ((ci & 63) << 3)]) = st[j];
  }
  __syncthreads();

  v8f racc = {};                               // per-lane partial row sums
  // B k order used here (contiguous, same as A): lane = column (lane&15), K half = (lane>=16)*16
  const int boff = (lane & 15) * LDSROW + ((lane >> 4) << 4);

  for (int nt = 0; nt < NTILES; ++nt) {
    const int cur = nt & 1;
    // issue next tile's global loads (latency overlapped with WMMAs below)
    if (nt + 1 < NTILES) {
      const _Float16* Fn = F + (size_t)(nt + 1) * 16 * H_;
#pragma unroll
      for (int j = 0; j < 4; ++j) {
        const int ci = tid + (j << 8);
        st[j] = *(const v8h*)(Fn + (size_t)(ci >> 6) * H_ + ((ci & 63) << 3));
      }
    }

    const _Float16* bb = &lds[cur][boff];
    v8f c0 = {}, c1 = {};
#pragma unroll
    for (int kt = 0; kt < KTILES; kt += 2) {   // two independent WMMA chains
      union { v16h v; v8h h[2]; } b0, b1;
      b0.h[0] = *(const v8h*)(bb + kt * 32);
      b0.h[1] = *(const v8h*)(bb + kt * 32 + 8);
      b1.h[0] = *(const v8h*)(bb + kt * 32 + 32);
      b1.h[1] = *(const v8h*)(bb + kt * 32 + 40);
      c0 = __builtin_amdgcn_wmma_f32_16x16x32_f16(
             false, a[kt].v,     false, b0.v, (short)0, c0, false, false);
      c1 = __builtin_amdgcn_wmma_f32_16x16x32_f16(
             false, a[kt + 1].v, false, b1.v, (short)0, c1, false, false);
    }

    float cr0 = c0[0] + c1[0];
#pragma unroll
    for (int v = 0; v < 8; ++v) racc[v] += fmaxf(c0[v] + c1[v], 0.0f);
    // +I on diagonal tiles (C layout: VGPR v = row v / v+8; lane&15 = col)
    if (mt == nt) {
#pragma unroll
      for (int v = 0; v < 8; ++v)
        if (lane == v || lane == v + 24) racc[v] += 1.0f;
    }
    // A0 = row 0 of clip(G,0)+I (m-tile 0 only; row 0 is VGPR0, lanes<16)
    if (mt == 0 && lane < 16) {
      const int ng = (nt << 4) + lane;
      a0row[(size_t)bm * NPAD + ng] = fmaxf(cr0, 0.0f) + (ng == 0 ? 1.0f : 0.0f);
    }

    // write staged tile into the other buffer, then sync
    if (nt + 1 < NTILES) {
#pragma unroll
      for (int j = 0; j < 4; ++j) {
        const int ci = tid + (j << 8);
        *(v8h*)(&lds[cur ^ 1][(ci >> 6) * LDSROW + ((ci & 63) << 3)]) = st[j];
      }
    }
    __syncthreads();
  }

  // reduce across the 16 lanes of each half (N runs over lanes)
#pragma unroll
  for (int off = 8; off >= 1; off >>= 1) {
#pragma unroll
    for (int v = 0; v < 8; ++v) racc[v] += __shfl_xor(racc[v], off, 32);
  }
  if (lane == 0 || lane == 16) {
    const int mbase = (mt << 4) + ((lane >> 4) << 3);
#pragma unroll
    for (int v = 0; v < 8; ++v)
      rowsum[(size_t)bm * NPAD + mbase + v] = racc[v];
  }
}

// ---------------------------------------------------------------------------
// K3: per (b,mod): adj_row0 = d0 * A0[m] * rsqrt(rowsum[m]);
//     v[h] = sum_m adj_row0[m] * feat[m][h]  (original f32 features)
//     g[k] = tanh(sum_h v[h]*W[h][k] + bias[0][k])
// ---------------------------------------------------------------------------
__global__ void __launch_bounds__(256)
k_graph(const float* __restrict__ in_text, const float* __restrict__ in_img,
        const float* __restrict__ base_text, const float* __restrict__ base_img,
        const float* __restrict__ W_tt, const float* __restrict__ b_tt,
        const float* __restrict__ W_it, const float* __restrict__ b_it,
        const float* __restrict__ rowsum, const float* __restrict__ a0row,
        float* __restrict__ gvec) {
  const int bm = blockIdx.x;
  const int b = bm >> 1, mod = bm & 1;
  const int t = threadIdx.x;
  __shared__ float adj0[N_];
  __shared__ float vsh[H_];

  const float* rs = rowsum + (size_t)bm * NPAD;
  const float* a0 = a0row + (size_t)bm * NPAD;
  const float d0 = rsqrtf(rs[0]);
  for (int m = t; m < N_; m += 256)
    adj0[m] = d0 * a0[m] * rsqrtf(rs[m]);
  __syncthreads();

  const float* inp  = (mod ? in_img : in_text) + (size_t)b * H_;
  const float* base = (mod ? base_img : base_text) + (size_t)b * C_ * H_;
  const int h0 = t, h1 = t + 256;
  float acc0 = adj0[0] * inp[h0];
  float acc1 = adj0[0] * inp[h1];
  for (int m = 1; m < N_; ++m) {
    const float w = adj0[m];
    const float* rp = base + (size_t)(m - 1) * H_;
    acc0 += w * rp[h0];
    acc1 += w * rp[h1];
  }
  vsh[h0] = acc0;
  vsh[h1] = acc1;
  __syncthreads();

  const float* W    = mod ? W_it : W_tt;
  const float* bias = mod ? b_it : b_tt;   // row 0 of [N,H] bias
  float g0 = bias[h0], g1 = bias[h1];
  for (int h = 0; h < H_; ++h) {
    const float vv = vsh[h];
    const float* wr = W + (size_t)h * H_;
    g0 += vv * wr[h0];
    g1 += vv * wr[h1];
  }
  gvec[(size_t)bm * H_ + h0] = tanhf(g0);
  gvec[(size_t)bm * H_ + h1] = tanhf(g1);
}

// ---------------------------------------------------------------------------
// K4: out_text = 0.5*base_text + 0.5*graph_o ; out_img likewise.
// graph_o = 0.7*g_tt + 0.3*g_it, broadcast over C.
// ---------------------------------------------------------------------------
__global__ void __launch_bounds__(256)
k_out(const float* __restrict__ base_text, const float* __restrict__ base_img,
      const float* __restrict__ gvec, float* __restrict__ out, long n) {
  long i = (long)blockIdx.x * blockDim.x + threadIdx.x;
  if (i >= n) return;
  const int k = (int)(i % H_);
  const int b = (int)(i / ((long)C_ * H_));
  const float gt = gvec[(size_t)(b * 2) * H_ + k];
  const float gi = gvec[(size_t)(b * 2 + 1) * H_ + k];
  const float go = 0.7f * gt + 0.3f * gi;
  out[i]     = 0.5f * base_text[i] + 0.5f * go;
  out[n + i] = 0.5f * base_img[i]  + 0.5f * go;
}

extern "C" void kernel_launch(void* const* d_in, const int* in_sizes, int n_in,
                              void* d_out, int out_size, void* d_ws, size_t ws_size,
                              hipStream_t stream) {
  const float* in_text   = (const float*)d_in[0];
  const float* in_img    = (const float*)d_in[1];
  const float* base_text = (const float*)d_in[2];
  const float* base_img  = (const float*)d_in[3];
  const float* W_tt      = (const float*)d_in[4];
  const float* b_tt      = (const float*)d_in[5];
  const float* W_it      = (const float*)d_in[6];
  const float* b_it      = (const float*)d_in[7];
  float* out = (float*)d_out;

  char* ws = (char*)d_ws;
  _Float16* Fh = (_Float16*)ws;                               // 64*1024*512 f16 = 67 MB
  size_t fhBytes = (size_t)64 * NPAD * H_ * sizeof(_Float16);
  float* rowsum = (float*)(ws + fhBytes);                     // 64*1024 f32
  float* a0row  = rowsum + 64 * NPAD;                         // 64*1024 f32
  float* gvec   = a0row + 64 * NPAD;                          // 64*512  f32

  k_normalize<<<dim3(64 * NPAD), dim3(256), 0, stream>>>(
      in_text, in_img, base_text, base_img, Fh);

  k_gram<<<dim3(NTILES / 8, 64), dim3(256), 0, stream>>>(Fh, rowsum, a0row);

  k_graph<<<dim3(64), dim3(256), 0, stream>>>(
      in_text, in_img, base_text, base_img,
      W_tt, b_tt, W_it, b_it, rowsum, a0row, gvec);

  const long n = (long)B_ * C_ * H_;
  k_out<<<dim3((unsigned)((n + 255) / 256)), dim3(256), 0, stream>>>(
      base_text, base_img, gvec, out, n);
}
